// GCN_TPOSE_66975720014540
// MI455X (gfx1250) — compile-verified, matrix operand order checked by reading, not run on hardware
//
#include <hip/hip_runtime.h>
#include <hip/hip_bf16.h>

// CDNA5 / gfx1250 implementation. Most matrix math lowers to
// v_wmma_f32_16x16x32_f16 via one generic batched *strided* GEMM kernel:
// every einsum in the model (GCN T/A einsums, pointwise convs, MLPs, iDCT)
// is expressed as (batch, M, N, K) + strides, avoiding transpose traffic.
// f16 operands / f32 accumulate (fp32 reference; WMMA f16 K=32 is the
// high-throughput path on MI455X — the f32 WMMA shape is only 16x16x4).
//
// v2: branch-free operand fetch. M/N bounds handled by clamping the source
// row/column (clamped-row garbage only reaches D elements the epilogue skips);
// K bounds handled in a separate tail chunk with clamped indices + cndmask
// zeroing. No EXEC-divergent loads -> loads issue as clauses, EXEC is all-ones
// at every WMMA by construction.

typedef __attribute__((ext_vector_type(16))) _Float16 v16h;
typedef __attribute__((ext_vector_type(8)))  float    v8f;
typedef long long i64;

#define BN_SC 0.9999950000375f   // 1/sqrt(1 + 1e-5)

struct GemmP {
  const _Float16* A;  i64 sAb, sAm, sAk;
  const _Float16* Bw; i64 sBb, sBk, sBn;
  float*    D;  i64 sDb,  sDm,  sDn;    // optional f32 out
  _Float16* Dh; i64 sDhb, sDhm, sDhn;   // optional f16 out
  const float* bias;                    // per-column
  const float* bnG; const float* bnB;   // batchnorm (folded eps)
  const float*    resid;  i64 sRb,  sRm,  sRn;   // f32 residual
  const _Float16* residH; i64 sRhb, sRhm, sRhn;  // f16 residual
  const float* prelu;                   // PReLU alpha (after resid)
  int actTanh;                          // tanh (before resid)
  int M, N, K;
};

__global__ __launch_bounds__(128) void gemm_wmma(GemmP p) {
  const int lane = threadIdx.x;
  const int hs   = lane >> 4;          // K-half select per ISA A/B layout
  const int l16  = lane & 15;
  const int n0   = blockIdx.x * 16;
  const int m0   = (blockIdx.y * 4 + threadIdx.y) * 16;
  const int b    = blockIdx.z;
  if (m0 >= p.M) return;               // wave-uniform: EXEC stays all-ones
  const int mr  = m0 + l16;
  const int nc  = n0 + l16;
  const int mrc = mr < p.M ? mr : p.M - 1;   // clamp; junk only reaches
  const int ncc = nc < p.N ? nc : p.N - 1;   // outputs we never store
  const _Float16* __restrict__ Ap = p.A  + (i64)b * p.sAb + (i64)mrc * p.sAm;
  const _Float16* __restrict__ Bp = p.Bw + (i64)b * p.sBb + (i64)ncc * p.sBn;

  v8f acc = {};
  int k0 = 0;
  {
    const _Float16* __restrict__ Ak = Ap + (i64)(hs * 8) * p.sAk;
    const _Float16* __restrict__ Bk = Bp + (i64)(hs * 8) * p.sBk;
    const i64 aStep = 32 * p.sAk, bStep = 32 * p.sBk;
    for (; k0 + 32 <= p.K; k0 += 32, Ak += aStep, Bk += bStep) {
      v16h av, bv;
#pragma unroll
      for (int i = 0; i < 8; ++i) {
        av[i]     = Ak[(i64)i * p.sAk];
        av[8 + i] = Ak[(i64)(16 + i) * p.sAk];
        bv[i]     = Bk[(i64)i * p.sBk];
        bv[8 + i] = Bk[(i64)(16 + i) * p.sBk];
      }
      acc = __builtin_amdgcn_wmma_f32_16x16x32_f16(false, av, false, bv,
                                                   (short)0, acc, false, false);
    }
  }
  if (k0 < p.K) {                      // K tail: clamp index, cndmask-zero
    const int kb  = k0 + hs * 8;
    const int Km1 = p.K - 1;
    const _Float16 z = (_Float16)0.0f;
    v16h av, bv;
#pragma unroll
    for (int i = 0; i < 8; ++i) {
      const int k1 = kb + i, k2 = kb + 16 + i;
      const int k1c = k1 <= Km1 ? k1 : Km1;
      const int k2c = k2 <= Km1 ? k2 : Km1;
      const _Float16 a1 = Ap[(i64)k1c * p.sAk];
      const _Float16 a2 = Ap[(i64)k2c * p.sAk];
      const _Float16 b1 = Bp[(i64)k1c * p.sBk];
      const _Float16 b2 = Bp[(i64)k2c * p.sBk];
      av[i]     = (k1 <= Km1) ? a1 : z;
      av[8 + i] = (k2 <= Km1) ? a2 : z;
      bv[i]     = (k1 <= Km1) ? b1 : z;
      bv[8 + i] = (k2 <= Km1) ? b2 : z;
    }
    acc = __builtin_amdgcn_wmma_f32_16x16x32_f16(false, av, false, bv,
                                                 (short)0, acc, false, false);
  }

  const float alpha = p.prelu ? p.prelu[0] : 0.0f;
#pragma unroll
  for (int r = 0; r < 8; ++r) {
    const int mm = m0 + hs * 8 + r;    // C/D layout: VGPR r <-> M = r + 8*hs
    const int nn = n0 + l16;
    if (mm >= p.M || nn >= p.N) continue;
    float v = acc[r];
    if (p.bias)    v += p.bias[nn];
    if (p.bnG)     v  = v * (p.bnG[nn] * BN_SC) + p.bnB[nn];
    if (p.actTanh) v  = tanhf(v);
    if (p.resid)   v += p.resid [(i64)b*p.sRb  + (i64)mm*p.sRm  + (i64)nn*p.sRn ];
    if (p.residH)  v += (float)p.residH[(i64)b*p.sRhb + (i64)mm*p.sRhm + (i64)nn*p.sRhn];
    if (p.prelu && v < 0.0f) v *= alpha;
    if (p.D)  p.D [(i64)b*p.sDb  + (i64)mm*p.sDm  + (i64)nn*p.sDn ] = v;
    if (p.Dh) p.Dh[(i64)b*p.sDhb + (i64)mm*p.sDhm + (i64)nn*p.sDhn] = (_Float16)v;
  }
}

// Generic 4-D strided copy / cast (handles permutes, slices, broadcasts).
template <typename TI, typename TO>
__global__ void k_copy4(const TI* __restrict__ in, i64 oi, i64 si0, i64 si1, i64 si2, i64 si3,
                        TO* __restrict__ out, i64 oo, i64 so0, i64 so1, i64 so2, i64 so3,
                        int d1, int d2, int d3, long total) {
  long idx = (long)blockIdx.x * blockDim.x + threadIdx.x;
  if (idx >= total) return;
  int i3 = (int)(idx % d3); long t = idx / d3;
  int i2 = (int)(t % d2); t /= d2;
  int i1 = (int)(t % d1); i64 i0 = t / d1;
  out[oo + i0*so0 + (i64)i1*so1 + (i64)i2*so2 + (i64)i3*so3] =
      (TO)in[oi + i0*si0 + (i64)i1*si1 + (i64)i2*si2 + (i64)i3*si3];
}

// rhd[b, d*3+c] = sum_t dct[d,t] * x_ext[t, b, root_idx[c]]  (t>=30 clamps to 29)
__global__ void k_rhd(const float* __restrict__ x, const float* __restrict__ dct,
                      const int* __restrict__ ridx,
                      float* __restrict__ rhd, _Float16* __restrict__ rhdh) {
  int idx = blockIdx.x * blockDim.x + threadIdx.x;
  if (idx >= 128 * 60 * 3) return;
  int c = idx % 3; int q = idx / 3; int d = q % 60; int b = q / 60;
  int col = ridx[c];
  float acc = 0.f;
  for (int t = 0; t < 90; ++t) {
    int ts = t < 30 ? t : 29;
    acc += dct[d * 90 + t] * x[(i64)ts * 8064 + b * 63 + col];
  }
  rhd[b * 180 + d * 3 + c]  = acc;
  rhdh[b * 180 + d * 3 + c] = (_Float16)acc;
}

// Scan step prologue: root-joint override of ylast + build (128,3,31,21) f16 input.
__global__ void k_build_xcat(const float* __restrict__ yall, const float* __restrict__ ylast,
                             const float* __restrict__ rtraj, const int* __restrict__ ridx,
                             int tstep, float* __restrict__ ylast_mod, _Float16* __restrict__ xcat) {
  int idx = blockIdx.x * blockDim.x + threadIdx.x;
  if (idx >= 128 * 3 * 31 * 21) return;
  int v = idx % 21; int r = idx / 21; int t = r % 31; r /= 31; int c = r % 3; int n = r / 3;
  float val;
  if (t < 30) {
    val = yall[((n * 3 + c) * 30 + t) * 21 + v];
  } else {
    int f = v * 3 + c;
    val = ylast[n * 63 + f];
    for (int j = 0; j < 3; ++j)
      if (f == ridx[j]) val = rtraj[n * 270 + (30 + tstep) * 3 + j];
    ylast_mod[n * 63 + f] = val;
  }
  xcat[idx] = (_Float16)val;
}

// yl[n, v*3+c] = sum_t d3[n,c,t,v]*wT[t] + bT + ylast_mod
__global__ void k_final(const float* __restrict__ d3, const float* __restrict__ wT,
                        const float* __restrict__ bT, const float* __restrict__ ylast_mod,
                        float* __restrict__ yout, float* __restrict__ ylast) {
  int idx = blockIdx.x * blockDim.x + threadIdx.x;
  if (idx >= 128 * 63) return;
  int f = idx % 63; int n = idx / 63;
  int v = f / 3, c = f % 3;
  float acc = bT[0];
  for (int t = 0; t < 31; ++t) acc += wT[t] * d3[((i64)(n * 3 + c) * 31 + t) * 21 + v];
  acc += ylast_mod[idx];
  yout[idx]  = acc;
  ylast[idx] = acc;
}

// yall_new = concat(yall_old[:,:,1:], yl)
__global__ void k_shift(const float* __restrict__ yall_old, const float* __restrict__ ylast,
                        float* __restrict__ yall_new) {
  int idx = blockIdx.x * blockDim.x + threadIdx.x;
  if (idx >= 128 * 3 * 30 * 21) return;
  int v = idx % 21; int r = idx / 21; int t = r % 30; r /= 30; int c = r % 3; int n = r / 3;
  yall_new[idx] = (t < 29) ? yall_old[((n * 3 + c) * 30 + t + 1) * 21 + v]
                           : ylast[n * 63 + v * 3 + c];
}

// ---------------- host orchestration ----------------

static inline unsigned cdivu(long a, long b) { return (unsigned)((a + b - 1) / b); }

static void gemm(hipStream_t st, const GemmP& p, int batch) {
  dim3 grid(cdivu(p.N, 16), cdivu(p.M, 64), (unsigned)batch);
  gemm_wmma<<<grid, dim3(32, 4, 1), 0, st>>>(p);
}

template <typename TI, typename TO>
static void copy4(hipStream_t st, const TI* in, i64 oi, i64 si0, i64 si1, i64 si2, i64 si3,
                  TO* out, i64 oo, i64 so0, i64 so1, i64 so2, i64 so3,
                  long d0, long d1, long d2, long d3) {
  long total = d0 * d1 * d2 * d3;
  k_copy4<TI, TO><<<cdivu(total, 256), 256, 0, st>>>(
      in, oi, si0, si1, si2, si3, out, oo, so0, so1, so2, so3,
      (int)d1, (int)d2, (int)d3, total);
}

static GemmP mk(const _Float16* A, i64 sAb, i64 sAm, i64 sAk,
                const _Float16* Bw, i64 sBb, i64 sBk, i64 sBn,
                int M, int N, int K) {
  GemmP p{};
  p.A = A; p.sAb = sAb; p.sAm = sAm; p.sAk = sAk;
  p.Bw = Bw; p.sBb = sBb; p.sBk = sBk; p.sBn = sBn;
  p.M = M; p.N = N; p.K = K;
  return p;
}

// setup_inputs() flattening order (depth-first, dict insertion order)
enum InIdx {
  IN_X = 0, IN_CONT, IN_CMASK, IN_DCT, IN_IDCT, IN_RIDX, IN_HOR,
  XCP_T, XCP_A, XCP_CW, XCP_CB, XCP_BG, XCP_BB, XCP_RW, XCP_RB, XCP_RG, XCP_RB2, XCP_PR,
  XT_W1, XT_B1, XT_W2, XT_B2,
  XN_W1, XN_B1, XN_W2, XN_B2,
  CCP_T, CCP_A, CCP_CW, CCP_CB, CCP_BG, CCP_BB, CCP_RW, CCP_RB, CCP_RG, CCP_RB2, CCP_PR,
  CT_W1, CT_B1, CT_W2, CT_B2,
  CN_W1, CN_B1, CN_W2, CN_B2,
  G1_W, G1_B,
  G2_W1, G2_B1, G2_W2, G2_B2,
  G3_W1, G3_B1, G3_W2, G3_B2,
  G4_W, G4_B,
  XD_T, XD_A, XD_CW, XD_CB, XD_BG, XD_BB, XD_RW, XD_RB, XD_RG, XD_RB2, XD_PR,
  D0_T, D0_A, D0_CW, D0_CB, D0_BG, D0_BB, D0_RW, D0_RB, D0_RG, D0_RB2, D0_PR,
  D1_T, D1_A, D1_CW, D1_CB, D1_BG, D1_BB, D1_RW, D1_RB, D1_RG, D1_RB2, D1_PR,
  DT_W, DT_B
};

extern "C" void kernel_launch(void* const* d_in, const int* in_sizes, int n_in,
                              void* d_out, int out_size, void* d_ws, size_t ws_size,
                              hipStream_t stream) {
  (void)in_sizes; (void)n_in; (void)out_size; (void)ws_size;
  auto F = [&](int i) { return (const float*)d_in[i]; };
  const int* ridx = (const int*)d_in[IN_RIDX];
  float* out_y  = (float*)d_out;                  // (60,128,63)
  float* out_rt = (float*)d_out + 60 * 128 * 63;  // (128,90,3)

  char* wsb = (char*)d_ws;
  size_t off = 0;
  auto allocB = [&](size_t bytes) { void* p = wsb + off; off += (bytes + 255) & ~(size_t)255; return p; };
  auto allocH = [&](size_t n) { return (_Float16*)allocB(n * 2); };
  auto allocF = [&](size_t n) { return (float*)allocB(n * 4); };
  auto castW  = [&](int idx, size_t n) {
    _Float16* dst = allocH(n);
    copy4<float, _Float16>(stream, F(idx), 0, 1, 0, 0, 0, dst, 0, 1, 0, 0, 0, (long)n, 1, 1, 1);
    return dst;
  };

  // ---- f16 weight shadows (recomputed each call: deterministic) ----
  _Float16* w_xT  = castW(XCP_T, 21*30*30);
  _Float16* w_xA  = castW(XCP_A, 30*21*21);
  _Float16* w_xcw = castW(XCP_CW, 256*3);
  _Float16* w_xrw = castW(XCP_RW, 256*3);
  _Float16* w_cT  = castW(CCP_T, 21*60*60);
  _Float16* w_cA  = castW(CCP_A, 60*21*21);
  _Float16* w_ccw = castW(CCP_CW, 256*4);
  _Float16* w_crw = castW(CCP_RW, 256*4);
  _Float16* w_xt1 = castW(XT_W1, 10*30);
  _Float16* w_xt2 = castW(XT_W2, 10);
  _Float16* w_xn1 = castW(XN_W1, 10*21);
  _Float16* w_xn2 = castW(XN_W2, 10);
  _Float16* w_ct1 = castW(CT_W1, 10*60);
  _Float16* w_ct2 = castW(CT_W2, 10);
  _Float16* w_cn1 = castW(CN_W1, 10*21);
  _Float16* w_cn2 = castW(CN_W2, 10);
  _Float16* w_g1  = castW(G1_W, 256*692);
  _Float16* w_g2a = castW(G2_W1, 256*256);
  _Float16* w_g2b = castW(G2_W2, 256*256);
  _Float16* w_g3a = castW(G3_W1, 256*256);
  _Float16* w_g3b = castW(G3_W2, 256*256);
  _Float16* w_g4  = castW(G4_W, 180*256);
  _Float16* w_dT  = castW(XD_T, 21*31*31);
  _Float16* w_dA  = castW(XD_A, 31*21*21);
  _Float16* w_dcw = castW(XD_CW, 256*3);
  _Float16* w_drw = castW(XD_RW, 256*3);
  _Float16* w_0T  = castW(D0_T, 21*31*31);
  _Float16* w_0A  = castW(D0_A, 31*21*21);
  _Float16* w_0cw = castW(D0_CW, 32*340);
  _Float16* w_0rw = castW(D0_RW, 32*340);
  _Float16* w_1T  = castW(D1_T, 21*31*31);
  _Float16* w_1A  = castW(D1_A, 31*21*21);
  _Float16* w_1cw = castW(D1_CW, 3*32);
  _Float16* w_1rw = castW(D1_RW, 3*32);
  _Float16* w_idct = castW(IN_IDCT, 90*60);

  // ---- workspace (reused across phases) ----
  _Float16* resH  = allocH(41287680);  // GCN residual (f16), max 128*256*1260
  _Float16* slotB = allocH(41287680);  // hxh / hch / dcat
  _Float16* slotC = allocH(28331520);  // T-einsum out
  _Float16* slotD = allocH(28331520);  // A-einsum out
  _Float16* xi_h  = allocH(241920);
  _Float16* ct_h  = allocH(645120);
  _Float16* t1buf = allocH(6881280);
  _Float16* h2buf = allocH(688128);
  _Float16* t2buf = allocH(327680);
  float*    hxCP  = allocF(32768);
  float*    hcCP  = allocF(32768);
  float*    rhd   = allocF(23040);
  _Float16* rhdh  = allocH(23040);
  _Float16* gcat  = allocH(88576);
  float*    g0    = allocF(32768);
  _Float16* g0h   = allocH(32768);
  _Float16* th_   = allocH(32768);
  float*    g1f   = allocF(32768);
  _Float16* g1h   = allocH(32768);
  _Float16* g2h   = allocH(32768);
  _Float16* rph   = allocH(23040);
  _Float16* xcat_h = allocH(249984);
  float*    ylast = allocF(8064);
  float*    ylmod = allocF(8064);
  float*    yall_a = allocF(241920);
  float*    yall_b = allocF(241920);
  _Float16* d32h  = allocH(2666496);
  float*    d3    = allocF(249984);

  // ---- build xi (n,c,t,v) from x (t,n,v*3+c); init scan state ----
  copy4<float,_Float16>(stream, F(IN_X),0, 63,1,8064,3, xi_h,0, 1890,630,21,1, 128,3,30,21);
  copy4<float,float>   (stream, F(IN_X),0, 63,1,8064,3, yall_a,0, 1890,630,21,1, 128,3,30,21);
  copy4<float,float>   (stream, F(IN_X),(i64)29*8064, 63,1,0,0, ylast,0, 63,1,0,0, 128,63,1,1);
  copy4<float,_Float16>(stream, F(IN_CONT),0, 5040,1,84,4, ct_h,0, 5040,1260,21,1, 128,4,60,21);

  // ---- x_CP GCN ----
  { GemmP p = mk(xi_h,1890,1,630, w_xrw,0,1,3, 630,256,3);
    p.bias=F(XCP_RB); p.bnG=F(XCP_RG); p.bnB=F(XCP_RB2);
    p.Dh=resH; p.sDhb=161280; p.sDhm=1; p.sDhn=630; gemm(stream,p,128); }
  { GemmP p = mk(xi_h,1,630,21, w_xT,900,30,1, 384,30,30);
    p.Dh=slotC; p.sDhb=1; p.sDhm=630; p.sDhn=21; gemm(stream,p,21); }
  { GemmP p = mk(slotC,21,630,1, w_xA,441,21,1, 384,21,21);
    p.Dh=slotD; p.sDhb=21; p.sDhm=630; p.sDhn=1; gemm(stream,p,30); }
  { GemmP p = mk(slotD,1890,1,630, w_xcw,0,1,3, 630,256,3);
    p.bias=F(XCP_CB); p.bnG=F(XCP_BG); p.bnB=F(XCP_BB);
    p.residH=resH; p.sRhb=161280; p.sRhm=1; p.sRhn=630; p.prelu=F(XCP_PR);
    p.Dh=slotB; p.sDhb=161280; p.sDhm=1; p.sDhn=630; gemm(stream,p,128); }
  // ---- x_time / x_node MLPs ----
  { GemmP p = mk(slotB,630,1,21, w_xt1,0,1,30, 21,10,30);
    p.bias=F(XT_B1); p.actTanh=1; p.Dh=t1buf; p.sDhb=210; p.sDhm=10; p.sDhn=1; gemm(stream,p,32768); }
  { GemmP p = mk(t1buf,210,10,1, w_xt2,0,1,10, 21,1,10);
    p.bias=F(XT_B2); p.actTanh=1; p.Dh=h2buf; p.sDhb=21; p.sDhm=1; p.sDhn=1; gemm(stream,p,32768); }
  { GemmP p = mk(h2buf,0,21,1, w_xn1,0,1,21, 32768,10,21);
    p.bias=F(XN_B1); p.actTanh=1; p.Dh=t2buf; p.sDhm=10; p.sDhn=1; gemm(stream,p,1); }
  { GemmP p = mk(t2buf,0,10,1, w_xn2,0,1,10, 32768,1,10);
    p.bias=F(XN_B2); p.actTanh=1; p.D=hxCP; p.sDm=1; p.sDn=1; gemm(stream,p,1); }
  copy4<float,_Float16>(stream, hxCP,0, 256,1,0,0, gcat,180, 692,1,0,0, 128,256,1,1);

  // ---- cont_CP GCN ----
  { GemmP p = mk(ct_h,5040,1,1260, w_crw,0,1,4, 1260,256,4);
    p.bias=F(CCP_RB); p.bnG=F(CCP_RG); p.bnB=F(CCP_RB2);
    p.Dh=resH; p.sDhb=322560; p.sDhm=1; p.sDhn=1260; gemm(stream,p,128); }
  { GemmP p = mk(ct_h,1,1260,21, w_cT,3600,60,1, 512,60,60);
    p.Dh=slotC; p.sDhb=1; p.sDhm=1260; p.sDhn=21; gemm(stream,p,21); }
  { GemmP p = mk(slotC,21,1260,1, w_cA,441,21,1, 512,21,21);
    p.Dh=slotD; p.sDhb=21; p.sDhm=1260; p.sDhn=1; gemm(stream,p,60); }
  { GemmP p = mk(slotD,5040,1,1260, w_ccw,0,1,4, 1260,256,4);
    p.bias=F(CCP_CB); p.bnG=F(CCP_BG); p.bnB=F(CCP_BB);
    p.residH=resH; p.sRhb=322560; p.sRhm=1; p.sRhn=1260; p.prelu=F(CCP_PR);
    p.Dh=slotB; p.sDhb=322560; p.sDhm=1; p.sDhn=1260; gemm(stream,p,128); }
  // ---- cont_time / cont_node ----
  { GemmP p = mk(slotB,1260,1,21, w_ct1,0,1,60, 21,10,60);
    p.bias=F(CT_B1); p.actTanh=1; p.Dh=t1buf; p.sDhb=210; p.sDhm=10; p.sDhn=1; gemm(stream,p,32768); }
  { GemmP p = mk(t1buf,210,10,1, w_ct2,0,1,10, 21,1,10);
    p.bias=F(CT_B2); p.actTanh=1; p.Dh=h2buf; p.sDhb=21; p.sDhm=1; p.sDhn=1; gemm(stream,p,32768); }
  { GemmP p = mk(h2buf,0,21,1, w_cn1,0,1,21, 32768,10,21);
    p.bias=F(CN_B1); p.actTanh=1; p.Dh=t2buf; p.sDhm=10; p.sDhn=1; gemm(stream,p,1); }
  { GemmP p = mk(t2buf,0,10,1, w_cn2,0,1,10, 32768,1,10);
    p.bias=F(CN_B2); p.actTanh=1; p.D=hcCP; p.sDm=1; p.sDn=1; gemm(stream,p,1); }
  copy4<float,_Float16>(stream, hcCP,0, 256,1,0,0, gcat,436, 692,1,0,0, 128,256,1,1);

  // ---- DCT head ----
  k_rhd<<<cdivu(128*60*3,256),256,0,stream>>>(F(IN_X), F(IN_DCT), ridx, rhd, rhdh);
  copy4<_Float16,_Float16>(stream, rhdh,0, 180,1,0,0, gcat,0, 692,1,0,0, 128,180,1,1);

  { GemmP p = mk(gcat,0,692,1, w_g1,0,1,692, 128,256,692);
    p.bias=F(G1_B); p.actTanh=1;
    p.D=g0; p.sDm=256; p.sDn=1; p.Dh=g0h; p.sDhm=256; p.sDhn=1; gemm(stream,p,1); }
  { GemmP p = mk(g0h,0,256,1, w_g2a,0,1,256, 128,256,256);
    p.bias=F(G2_B1); p.actTanh=1; p.Dh=th_; p.sDhm=256; p.sDhn=1; gemm(stream,p,1); }
  { GemmP p = mk(th_,0,256,1, w_g2b,0,1,256, 128,256,256);
    p.bias=F(G2_B2); p.actTanh=1; p.resid=g0; p.sRm=256; p.sRn=1;
    p.D=g1f; p.sDm=256; p.sDn=1; p.Dh=g1h; p.sDhm=256; p.sDhn=1; gemm(stream,p,1); }
  { GemmP p = mk(g1h,0,256,1, w_g3a,0,1,256, 128,256,256);
    p.bias=F(G3_B1); p.actTanh=1; p.Dh=th_; p.sDhm=256; p.sDhn=1; gemm(stream,p,1); }
  { GemmP p = mk(th_,0,256,1, w_g3b,0,1,256, 128,256,256);
    p.bias=F(G3_B2); p.actTanh=1; p.resid=g1f; p.sRm=256; p.sRn=1;
    p.Dh=g2h; p.sDhm=256; p.sDhn=1; gemm(stream,p,1); }
  { GemmP p = mk(g2h,0,256,1, w_g4,0,1,256, 128,180,256);
    p.bias=F(G4_B); p.resid=rhd; p.sRm=180; p.sRn=1;
    p.Dh=rph; p.sDhm=180; p.sDhn=1; gemm(stream,p,1); }
  // root_traj[b,t,c] = sum_d idct[t,d] * root_pred[b, d*3+c]  (batched over c)
  { GemmP p = mk(rph,1,180,3, w_idct,0,1,60, 128,90,60);
    p.D=out_rt; p.sDb=1; p.sDm=270; p.sDn=3; gemm(stream,p,3); }

  // ---- 60-step decoder scan ----
  for (int t = 0; t < 60; ++t) {
    const float* ysrc = (t & 1) ? yall_b : yall_a;
    float*       ydst = (t & 1) ? yall_a : yall_b;
    k_build_xcat<<<cdivu(128*3*31*21,256),256,0,stream>>>(ysrc, ylast, out_rt, ridx, t, ylmod, xcat_h);
    // x_dec GCN (3 -> 256)
    { GemmP p = mk(xcat_h,1953,1,651, w_drw,0,1,3, 651,256,3);
      p.bias=F(XD_RB); p.bnG=F(XD_RG); p.bnB=F(XD_RB2);
      p.Dh=resH; p.sDhb=166656; p.sDhm=1; p.sDhn=651; gemm(stream,p,128); }
    { GemmP p = mk(xcat_h,1,651,21, w_dT,961,31,1, 384,31,31);
      p.Dh=slotC; p.sDhb=1; p.sDhm=651; p.sDhn=21; gemm(stream,p,21); }
    { GemmP p = mk(slotC,21,651,1, w_dA,441,21,1, 384,21,21);
      p.Dh=slotD; p.sDhb=21; p.sDhm=651; p.sDhn=1; gemm(stream,p,31); }
    { GemmP p = mk(slotD,1953,1,651, w_dcw,0,1,3, 651,256,3);
      p.bias=F(XD_CB); p.bnG=F(XD_BG); p.bnB=F(XD_BB);
      p.residH=resH; p.sRhb=166656; p.sRhm=1; p.sRhn=651; p.prelu=F(XD_PR);
      p.Dh=slotB; p.sDhb=221340; p.sDhm=1; p.sDhn=651; gemm(stream,p,128); }
    // context channels 256..339 broadcast over (t,v)
    copy4<float,_Float16>(stream, F(IN_CONT),(i64)t*84, 5040,1,0,0,
                          slotB,166656, 221340,651,1,0, 128,84,651,1);
    // dec0 GCN (340 -> 32)
    { GemmP p = mk(slotB,221340,1,651, w_0rw,0,1,340, 651,32,340);
      p.bias=F(D0_RB); p.bnG=F(D0_RG); p.bnB=F(D0_RB2);
      p.Dh=resH; p.sDhb=20832; p.sDhm=1; p.sDhn=651; gemm(stream,p,128); }
    { GemmP p = mk(slotB,1,651,21, w_0T,961,31,1, 43520,31,31);
      p.Dh=slotC; p.sDhb=1; p.sDhm=651; p.sDhn=21; gemm(stream,p,21); }
    { GemmP p = mk(slotC,21,651,1, w_0A,441,21,1, 43520,21,21);
      p.Dh=slotD; p.sDhb=21; p.sDhm=651; p.sDhn=1; gemm(stream,p,31); }
    { GemmP p = mk(slotD,221340,1,651, w_0cw,0,1,340, 651,32,340);
      p.bias=F(D0_CB); p.bnG=F(D0_BG); p.bnB=F(D0_BB);
      p.residH=resH; p.sRhb=20832; p.sRhm=1; p.sRhn=651; p.prelu=F(D0_PR);
      p.Dh=d32h; p.sDhb=20832; p.sDhm=1; p.sDhn=651; gemm(stream,p,128); }
    // dec1 GCN (32 -> 3)
    { GemmP p = mk(d32h,20832,1,651, w_1rw,0,1,32, 651,3,32);
      p.bias=F(D1_RB); p.bnG=F(D1_RG); p.bnB=F(D1_RB2);
      p.Dh=resH; p.sDhb=1953; p.sDhm=1; p.sDhn=651; gemm(stream,p,128); }
    { GemmP p = mk(d32h,1,651,21, w_1T,961,31,1, 4096,31,31);
      p.Dh=slotC; p.sDhb=1; p.sDhm=651; p.sDhn=21; gemm(stream,p,21); }
    { GemmP p = mk(slotC,21,651,1, w_1A,441,21,1, 4096,21,21);
      p.Dh=slotD; p.sDhb=21; p.sDhm=651; p.sDhn=1; gemm(stream,p,31); }
    { GemmP p = mk(slotD,20832,1,651, w_1cw,0,1,32, 651,3,32);
      p.bias=F(D1_CB); p.bnG=F(D1_BG); p.bnB=F(D1_BB);
      p.residH=resH; p.sRhb=1953; p.sRhm=1; p.sRhn=651; p.prelu=F(D1_PR);
      p.D=d3; p.sDb=1953; p.sDm=1; p.sDn=651; gemm(stream,p,128); }
    k_final<<<cdivu(128*63,256),256,0,stream>>>(d3, F(DT_W), F(DT_B), ylmod,
                                                out_y + (i64)t * 8064, ylast);
    k_shift<<<cdivu(128*3*30*21,256),256,0,stream>>>(ysrc, ylast, ydst);
  }
}
